// AGRNNCell_13211319403249
// MI455X (gfx1250) — matrix operand order repaired and checked by reading, not compiled, not run on hardware
//
#include <hip/hip_runtime.h>
#include <hip/hip_bf16.h>

#define B_   8
#define N_   1024
#define H_   4
#define KTOP 8
#define DH_  64
#define DM_  128
#define DFF_ 512
#define NT_  (B_ * N_)            // 8192
#define NE_  (NT_ * H_ * KTOP)    // 262144 undirected edge records

typedef __attribute__((ext_vector_type(16))) __bf16 v16bf;
typedef __attribute__((ext_vector_type(8)))  float  v8f;

union Frag { uint4 u[2]; v16bf v; };

// ---------------- CDNA5 async global->LDS copy (16B per lane) ---------------
__device__ __forceinline__ void async_copy16(const void* gptr, void* lptr) {
  unsigned int lds = (unsigned int)(unsigned long long)lptr;   // LDS offset = addr[31:0]
  asm volatile("global_load_async_to_lds_b128 %0, %1, off"
               :: "v"(lds), "v"((unsigned long long)gptr) : "memory");
}
__device__ __forceinline__ void wait_async0() {
  asm volatile("s_wait_asynccnt 0x0" ::: "memory");
}

// A (16x32 bf16, row-major src): lane r holds row M=r.
// chunk0 = K[16*half .. 16*half+7], chunk1 = K[16*half+8 .. 16*half+15]
__device__ __forceinline__ v16bf load_a_bf16(const __bf16* __restrict__ row, int k0, int half) {
  const uint4* p = (const uint4*)row;          // 8 bf16 per uint4
  Frag f;
  f.u[0] = p[(k0 >> 3) + 2 * half];
  f.u[1] = p[(k0 >> 3) + 2 * half + 1];
  return f.v;
}

// ---------------- WMMA GEMM: C = A[M,Kd]bf16 @ Wt[Nn,Kd]bf16 ----------------
// B tile staged through LDS with async copies, double buffered.
template <bool BIAS, bool RELU, bool OUTBF>
__global__ void k_gemm(const __bf16* __restrict__ A, const __bf16* __restrict__ Wt,
                       const float* __restrict__ bias, float* __restrict__ Cf,
                       __bf16* __restrict__ Cb, int Kd, int Nn) {
  __shared__ __align__(16) __bf16 btile[2][16][32];
  int lane = threadIdx.x & 31;
  int half = lane >> 4, r = lane & 15;
  int n0 = blockIdx.x * 16;
  int m0 = blockIdx.y * 16;
  const __bf16* arow = A + (size_t)(m0 + r) * Kd;
  const __bf16* wrow = Wt + (size_t)(n0 + r) * Kd + 16 * half;  // this lane's 16 K-values

  // preload first weight tile (k0 = 0)
  async_copy16(wrow,     &btile[0][r][16 * half]);
  async_copy16(wrow + 8, &btile[0][r][16 * half + 8]);

  v8f acc = {0.f, 0.f, 0.f, 0.f, 0.f, 0.f, 0.f, 0.f};
  int buf = 0;
  for (int k0 = 0; k0 < Kd; k0 += 32) {
    wait_async0();                          // LDS tile[buf] resident
    __builtin_amdgcn_wave_barrier();        // single-wave block: compiler fence only
    // B fragment: lane col = n0+r -> 16 contiguous bf16 from LDS (2x ds_load_b128)
    Frag fb;
    const uint4* lp = (const uint4*)&btile[buf][r][16 * half];
    fb.u[0] = lp[0];
    fb.u[1] = lp[1];
    // prefetch next weight tile into the other buffer while we compute
    if (k0 + 32 < Kd) {
      const __bf16* wn = wrow + k0 + 32;
      async_copy16(wn,     &btile[buf ^ 1][r][16 * half]);
      async_copy16(wn + 8, &btile[buf ^ 1][r][16 * half + 8]);
    }
    __builtin_prefetch(arow + k0 + 64, 0, 1);     // global_prefetch_b8
    v16bf fa = load_a_bf16(arow, k0, half);
    acc = __builtin_amdgcn_wmma_f32_16x16x32_bf16(false, fa, false, fb.v,
                                                  (short)0, acc, false, false);
    buf ^= 1;
  }
#pragma unroll
  for (int i = 0; i < 8; ++i) {
    int m = m0 + i + 8 * half;
    float v = acc[i];
    if (BIAS) v += bias[n0 + r];
    if (RELU) v = fmaxf(v, 0.f);
    if (OUTBF) Cb[(size_t)m * Nn + n0 + r] = (__bf16)v;
    else       Cf[(size_t)m * Nn + n0 + r] = v;
  }
}

// ---------------- weight pack: Wt[n][k] = bf16(W[k][n]) ---------------------
__global__ void k_packWt(const float* __restrict__ W, __bf16* __restrict__ Wt,
                         int Kd, int Nn) {
  int gid = blockIdx.x * blockDim.x + threadIdx.x;
  if (gid >= Kd * Nn) return;
  int n = gid / Kd, k = gid - n * Kd;
  Wt[gid] = (__bf16)W[(size_t)k * Nn + n];
}

// ---------------- s = [x @ W_in + b_in , state]  (f32 + bf16 copies) --------
__global__ void k_build_s(const float* __restrict__ x, const float* __restrict__ st,
                          const float* __restrict__ Win, const float* __restrict__ bin,
                          float* __restrict__ s, __bf16* __restrict__ sb) {
  int gid = blockIdx.x * blockDim.x + threadIdx.x;
  if (gid >= NT_ * DM_) return;
  int row = gid >> 7, c = gid & 127;
  float val;
  if (c < DH_) {
    float acc = bin[c];
#pragma unroll
    for (int i = 0; i < 16; ++i) acc += x[row * 16 + i] * Win[i * DH_ + c];
    val = acc;
  } else {
    val = st[row * DH_ + (c - DH_)];
  }
  s[gid] = val;
  sb[gid] = (__bf16)val;
}

// ---------------- streaming attention: WMMA scores + online softmax + top-8 -
__global__ void k_attn_topk(const __bf16* __restrict__ q, const __bf16* __restrict__ kmat,
                            float* __restrict__ Aout, int* __restrict__ idxout) {
  int lane = threadIdx.x & 31;
  int half = lane >> 4, r = lane & 15;
  int qt = blockIdx.x, h = blockIdx.y, b = blockIdx.z;
  int m0 = qt * 16;
  size_t base = (size_t)b * N_ * DM_;

  // Q fragment: 16 contiguous bf16 (K = 16*half..16*half+15) within the head slice
  Frag fq;
  {
    const uint4* qp = (const uint4*)(q + base + (size_t)(m0 + r) * DM_ + h * 32);
    fq.u[0] = qp[2 * half];
    fq.u[1] = qp[2 * half + 1];
  }

  __shared__ float tile[16][17];
  float m_run = -1e30f, l_run = 0.f;
  float ts[KTOP]; int ti[KTOP];
#pragma unroll
  for (int z = 0; z < KTOP; ++z) { ts[z] = -1e30f; ti[z] = 0; }
  const float scale = 0.17677669529663687f;   // 1/sqrt(32)

  // software pipeline: prefetch key tile kt+1 while scoring/scanning tile kt
  const __bf16* kbase = kmat + base + (size_t)r * DM_ + h * 32 + 16 * half;
  Frag fk;
  {
    const uint4* kp = (const uint4*)kbase;
    fk.u[0] = kp[0]; fk.u[1] = kp[1];
  }
  for (int kt = 0; kt < N_ / 16; ++kt) {
    Frag cur = fk;
    if (kt + 1 < N_ / 16) {
      const uint4* kp = (const uint4*)(kbase + (size_t)(kt + 1) * 16 * DM_);
      fk.u[0] = kp[0]; fk.u[1] = kp[1];
    }
    v8f sc = {0.f, 0.f, 0.f, 0.f, 0.f, 0.f, 0.f, 0.f};
    sc = __builtin_amdgcn_wmma_f32_16x16x32_bf16(false, fq.v, false, cur.v,
                                                 (short)0, sc, false, false);
#pragma unroll
    for (int i = 0; i < 8; ++i) tile[i + 8 * half][r] = sc[i] * scale;
    __builtin_amdgcn_wave_barrier();          // single-wave: DS stays in order
    if (lane < 16) {
#pragma unroll
      for (int c = 0; c < 16; ++c) {
        float sv = tile[lane][c];
        int key = kt * 16 + c;
        float mn = fmaxf(m_run, sv);
        l_run = l_run * __expf(m_run - mn) + __expf(sv - mn);
        m_run = mn;
        int amin = 0; float vmin = ts[0];
#pragma unroll
        for (int z = 1; z < KTOP; ++z) if (ts[z] < vmin) { vmin = ts[z]; amin = z; }
        if (sv > vmin) { ts[amin] = sv; ti[amin] = key; }
      }
    }
    __builtin_amdgcn_wave_barrier();
  }

  if (lane < 16) {
    // sort descending (stable by index) to match jax.lax.top_k
#pragma unroll
    for (int a = 0; a < KTOP; ++a) {
      int best = a;
#pragma unroll
      for (int z = a + 1; z < KTOP; ++z)
        if (ts[z] > ts[best] || (ts[z] == ts[best] && ti[z] < ti[best])) best = z;
      float tv = ts[a]; ts[a] = ts[best]; ts[best] = tv;
      int   tj = ti[a]; ti[a] = ti[best]; ti[best] = tj;
    }
    int n = m0 + lane;
    size_t o = ((((size_t)b * H_ + h) * N_) + n) * KTOP;
    float inv = 1.0f / l_run;
#pragma unroll
    for (int z = 0; z < KTOP; ++z) {
      Aout[o + z]   = __expf(ts[z] - m_run) * inv;
      idxout[o + z] = ti[z];
    }
  }
}

// ---------------- gather: attn[b,n,h*32+d] = sum_k A_k * v[b,idx_k,h*32+d] --
__global__ void k_attn_gather(const __bf16* __restrict__ v, const float* __restrict__ Aout,
                              const int* __restrict__ idxout, __bf16* __restrict__ attn) {
  int lane = threadIdx.x & 31;
  int n = blockIdx.x & (N_ - 1);
  int h = (blockIdx.x >> 10) & (H_ - 1);
  int b = blockIdx.x >> 12;
  size_t o = ((((size_t)b * H_ + h) * N_) + n) * KTOP;
  float acc = 0.f;
#pragma unroll
  for (int z = 0; z < KTOP; ++z) {
    int key = idxout[o + z];
    acc += Aout[o + z] * (float)v[((size_t)b * N_ + key) * DM_ + h * 32 + lane];
  }
  attn[((size_t)b * N_ + n) * DM_ + h * 32 + lane] = (__bf16)acc;
}

// ---------------- out = LayerNorm(xa + xb)  (f32 + optional bf16) -----------
__global__ void k_add_ln(const float* __restrict__ xa, const float* __restrict__ xb,
                         const float* __restrict__ g, const float* __restrict__ bt,
                         float* __restrict__ out, __bf16* __restrict__ outb) {
  int row = blockIdx.x;
  int lane = threadIdx.x & 31;
  float v[4]; float s = 0.f;
#pragma unroll
  for (int j = 0; j < 4; ++j) {
    int c = lane + 32 * j;
    v[j] = xa[(size_t)row * DM_ + c] + xb[(size_t)row * DM_ + c];
    s += v[j];
  }
#pragma unroll
  for (int off = 16; off; off >>= 1) s += __shfl_xor(s, off, 32);
  float mean = s * (1.0f / DM_);
  float var = 0.f;
#pragma unroll
  for (int j = 0; j < 4; ++j) { float d = v[j] - mean; var += d * d; }
#pragma unroll
  for (int off = 16; off; off >>= 1) var += __shfl_xor(var, off, 32);
  float rinv = rsqrtf(var * (1.0f / DM_) + 1e-5f);
#pragma unroll
  for (int j = 0; j < 4; ++j) {
    int c = lane + 32 * j;
    float o = (v[j] - mean) * rinv * g[c] + bt[c];
    out[(size_t)row * DM_ + c] = o;
    if (outb) outb[(size_t)row * DM_ + c] = (__bf16)o;
  }
}

// ---------------- graph emission + degrees ----------------------------------
__global__ void k_zero(float* __restrict__ p, int n) {
  int gid = blockIdx.x * blockDim.x + threadIdx.x;
  if (gid < n) p[gid] = 0.f;
}

__global__ void k_edges(const float* __restrict__ Aout, const int* __restrict__ idxout,
                        float* __restrict__ o_src, float* __restrict__ o_dst,
                        float* __restrict__ o_ew, int* __restrict__ esrc,
                        int* __restrict__ edst, float* __restrict__ ewt,
                        float* __restrict__ deg) {
  int e = blockIdx.x * blockDim.x + threadIdx.x;
  if (e >= NE_) return;
  int k_ = e & 7;
  int h  = (e >> 3) & 3;
  int n  = (e >> 5) & (N_ - 1);
  int b  = e >> 15;
  int s1 = b * N_ + n;
  size_t ai = ((((size_t)b * H_ + h) * N_) + n) * KTOP + k_;
  int d1 = idxout[ai] + b * N_;
  float w = Aout[ai];
  esrc[e] = s1; edst[e] = d1; ewt[e] = w;
  o_src[e] = (float)s1; o_src[NE_ + e] = (float)d1;
  o_dst[e] = (float)d1; o_dst[NE_ + e] = (float)s1;
  o_ew[e]  = w;         o_ew[NE_ + e]  = w;
  atomicAdd(&deg[s1], w);
  atomicAdd(&deg[d1], w);
}

__global__ void k_dinv(const float* __restrict__ deg, float* __restrict__ dinv) {
  int i = blockIdx.x * blockDim.x + threadIdx.x;
  if (i < NT_) dinv[i] = rsqrtf(deg[i] + 1.0f);   // +1 = self loop
}

__global__ void k_gcn_init(float* __restrict__ out, const float* __restrict__ bias,
                           const float* __restrict__ dinv, const float* __restrict__ hf) {
  int gid = blockIdx.x * blockDim.x + threadIdx.x;
  if (gid >= NT_ * DH_) return;
  int i = gid >> 6, f = gid & 63;
  float di = dinv[i];
  out[gid] = bias[f] + di * di * hf[gid];
}

// scatter over 2*NE_ directed edges; each thread handles 4 features (float4)
__global__ void k_gcn_scatter(float* __restrict__ out, const float* __restrict__ hf,
                              const int* __restrict__ esrc, const int* __restrict__ edst,
                              const float* __restrict__ ewt, const float* __restrict__ dinv) {
  long long gid = (long long)blockIdx.x * blockDim.x + threadIdx.x;
  if (gid >= (long long)2 * NE_ * (DH_ / 4)) return;
  int f4 = (int)(gid & 15);               // 16 groups of 4 features
  int e2 = (int)(gid >> 4);
  int s, d; float w;
  if (e2 < NE_) { s = esrc[e2]; d = edst[e2]; w = ewt[e2]; }
  else          { s = edst[e2 - NE_]; d = esrc[e2 - NE_]; w = ewt[e2 - NE_]; }
  float coef = dinv[s] * w * dinv[d];
  const float4 hv = *(const float4*)&hf[(size_t)s * DH_ + 4 * f4];
  float* op = &out[(size_t)d * DH_ + 4 * f4];
  atomicAdd(op + 0, coef * hv.x);
  atomicAdd(op + 1, coef * hv.y);
  atomicAdd(op + 2, coef * hv.z);
  atomicAdd(op + 3, coef * hv.w);
}

__global__ void k_sigmoid(float* __restrict__ g, int n) {
  int gid = blockIdx.x * blockDim.x + threadIdx.x;
  if (gid < n) g[gid] = 1.0f / (1.0f + __expf(-g[gid]));
}

// cat_bf16 = [xx , zg * state]
__global__ void k_cat(__bf16* __restrict__ cat, const float* __restrict__ xx,
                      const float* __restrict__ zg, const float* __restrict__ st) {
  int gid = blockIdx.x * blockDim.x + threadIdx.x;
  if (gid >= NT_ * 192) return;
  int i = gid / 192, c = gid - i * 192;
  float v = (c < DM_) ? xx[(size_t)i * DM_ + c]
                      : zg[(size_t)i * DH_ + (c - DM_)] * st[(size_t)i * DH_ + (c - DM_)];
  cat[gid] = (__bf16)v;
}

__global__ void k_final(float* __restrict__ out, const float* __restrict__ rg,
                        const float* __restrict__ st, const float* __restrict__ gu) {
  int gid = blockIdx.x * blockDim.x + threadIdx.x;
  if (gid >= NT_ * DH_) return;
  float r = rg[gid];
  out[gid] = r * st[gid] + (1.0f - r) * tanhf(gu[gid]);
}

// ---------------------------------------------------------------------------
extern "C" void kernel_launch(void* const* d_in, const int* in_sizes, int n_in,
                              void* d_out, int out_size, void* d_ws, size_t ws_size,
                              hipStream_t stream) {
  const float* x     = (const float*)d_in[0];
  const float* state = (const float*)d_in[1];
  const float* W_in  = (const float*)d_in[2];
  const float* b_in  = (const float*)d_in[3];
  const float* Wq = (const float*)d_in[4];  const float* bq = (const float*)d_in[5];
  const float* Wk = (const float*)d_in[6];  const float* bk = (const float*)d_in[7];
  const float* Wv = (const float*)d_in[8];  const float* bv = (const float*)d_in[9];
  const float* Wo = (const float*)d_in[10]; const float* bo = (const float*)d_in[11];
  const float* ln1g = (const float*)d_in[12]; const float* ln1b = (const float*)d_in[13];
  const float* W1 = (const float*)d_in[14]; const float* b1 = (const float*)d_in[15];
  const float* W2 = (const float*)d_in[16]; const float* b2 = (const float*)d_in[17];
  const float* ln2g = (const float*)d_in[18]; const float* ln2b = (const float*)d_in[19];
  const float* Wz = (const float*)d_in[20]; const float* bz = (const float*)d_in[21];
  const float* Wr = (const float*)d_in[22]; const float* br = (const float*)d_in[23];
  const float* Wu = (const float*)d_in[24]; const float* bu = (const float*)d_in[25];

  // output layout: h[524288] | src[524288] | dst[524288] | ew[524288]
  float* out_h   = (float*)d_out;
  float* out_src = out_h + (size_t)NT_ * DH_;
  float* out_dst = out_src + 2 * (size_t)NE_;
  float* out_ew  = out_dst + 2 * (size_t)NE_;

  // workspace carve-up (64B aligned blocks)
  float* w = (float*)d_ws;
  size_t off = 0;
  auto allocF = [&](size_t n) { off = (off + 15) & ~(size_t)15; float* p = w + off; off += n; return p; };
  auto allocB = [&](size_t n_bf) { return (__bf16*)allocF((n_bf + 1) / 2); };

  float*  s_    = allocF((size_t)NT_ * DM_);
  float*  atno_ = allocF((size_t)NT_ * DM_);
  float*  y_    = allocF((size_t)NT_ * DM_);
  float*  ff2_  = allocF((size_t)NT_ * DM_);
  float*  xx_   = allocF((size_t)NT_ * DM_);
  __bf16* s_b   = allocB((size_t)NT_ * DM_);
  __bf16* q_b   = allocB((size_t)NT_ * DM_);
  __bf16* k_b   = allocB((size_t)NT_ * DM_);
  __bf16* v_b   = allocB((size_t)NT_ * DM_);
  __bf16* at_b  = allocB((size_t)NT_ * DM_);
  __bf16* y_b   = allocB((size_t)NT_ * DM_);
  __bf16* xx_b  = allocB((size_t)NT_ * DM_);
  __bf16* ff1_b = allocB((size_t)NT_ * DFF_);
  __bf16* cat_b = allocB((size_t)NT_ * 192);
  __bf16* Wq_t = allocB((size_t)DM_ * DM_);
  __bf16* Wk_t = allocB((size_t)DM_ * DM_);
  __bf16* Wv_t = allocB((size_t)DM_ * DM_);
  __bf16* Wo_t = allocB((size_t)DM_ * DM_);
  __bf16* W1_t = allocB((size_t)DM_ * DFF_);
  __bf16* W2_t = allocB((size_t)DFF_ * DM_);
  __bf16* Wz_t = allocB((size_t)DM_ * DH_);
  __bf16* Wr_t = allocB((size_t)DM_ * DH_);
  __bf16* Wu_t = allocB((size_t)192 * DH_);
  float* Abuf = allocF(NE_);
  int*   idxb = (int*)allocF(NE_);
  int*   esrc = (int*)allocF(NE_);
  int*   edst = (int*)allocF(NE_);
  float* ewt  = allocF(NE_);
  float* deg  = allocF(NT_);
  float* dinv = allocF(NT_);
  float* hz_  = allocF((size_t)NT_ * DH_);
  float* hr_  = allocF((size_t)NT_ * DH_);
  float* hu_  = allocF((size_t)NT_ * DH_);
  float* gz_  = allocF((size_t)NT_ * DH_);
  float* gr_  = allocF((size_t)NT_ * DH_);
  float* gu_  = allocF((size_t)NT_ * DH_);
  (void)ws_size; (void)in_sizes; (void)n_in; (void)out_size;

  // 0. pack weights to bf16, transposed [N][K]
  auto pack = [&](const float* W, __bf16* Wt, int Kd, int Nn) {
    k_packWt<<<(Kd * Nn + 255) / 256, 256, 0, stream>>>(W, Wt, Kd, Nn);
  };
  pack(Wq, Wq_t, DM_, DM_);  pack(Wk, Wk_t, DM_, DM_);
  pack(Wv, Wv_t, DM_, DM_);  pack(Wo, Wo_t, DM_, DM_);
  pack(W1, W1_t, DM_, DFF_); pack(W2, W2_t, DFF_, DM_);
  pack(Wz, Wz_t, DM_, DH_);  pack(Wr, Wr_t, DM_, DH_);
  pack(Wu, Wu_t, 192, DH_);

  // 1. s = [x@W_in + b_in, state]
  k_build_s<<<(NT_ * DM_ + 255) / 256, 256, 0, stream>>>(x, state, W_in, b_in, s_, s_b);

  // 2. QKV projections (WMMA, async-LDS weight staging, bf16 out)
  dim3 g128(DM_ / 16, NT_ / 16);
  k_gemm<true, false, true><<<g128, 32, 0, stream>>>(s_b, Wq_t, bq, nullptr, q_b, DM_, DM_);
  k_gemm<true, false, true><<<g128, 32, 0, stream>>>(s_b, Wk_t, bk, nullptr, k_b, DM_, DM_);
  k_gemm<true, false, true><<<g128, 32, 0, stream>>>(s_b, Wv_t, bv, nullptr, v_b, DM_, DM_);

  // 3. streaming attention scores + online softmax + top-8
  k_attn_topk<<<dim3(N_ / 16, H_, B_), 32, 0, stream>>>(q_b, k_b, Abuf, idxb);

  // 4. weighted gather of V (bf16 out feeds Wo GEMM)
  k_attn_gather<<<B_ * H_ * N_, 32, 0, stream>>>(v_b, Abuf, idxb, at_b);

  // 5. output projection + residual LN (y kept in f32 + bf16)
  k_gemm<true, false, false><<<g128, 32, 0, stream>>>(at_b, Wo_t, bo, atno_, nullptr, DM_, DM_);
  k_add_ln<<<NT_, 32, 0, stream>>>(s_, atno_, ln1g, ln1b, y_, y_b);

  // 6. FFN + residual LN
  k_gemm<true, true, true><<<dim3(DFF_ / 16, NT_ / 16), 32, 0, stream>>>(y_b, W1_t, b1, nullptr, ff1_b, DM_, DFF_);
  k_gemm<true, false, false><<<dim3(DM_ / 16, NT_ / 16), 32, 0, stream>>>(ff1_b, W2_t, b2, ff2_, nullptr, DFF_, DM_);
  k_add_ln<<<NT_, 32, 0, stream>>>(y_, ff2_, ln2g, ln2b, xx_, xx_b);

  // 7. dynamic graph: edges, weights, degrees
  k_zero<<<(NT_ + 255) / 256, 256, 0, stream>>>(deg, NT_);
  k_edges<<<(NE_ + 255) / 256, 256, 0, stream>>>(Abuf, idxb, out_src, out_dst, out_ew,
                                                 esrc, edst, ewt, deg);
  k_dinv<<<(NT_ + 255) / 256, 256, 0, stream>>>(deg, dinv);

  // 8. GCN dense parts (WMMA) + sparse scatter (atomics, float4 per thread)
  dim3 g64(DH_ / 16, NT_ / 16);
  k_gemm<false, false, false><<<g64, 32, 0, stream>>>(xx_b, Wz_t, nullptr, hz_, nullptr, DM_, DH_);
  k_gemm<false, false, false><<<g64, 32, 0, stream>>>(xx_b, Wr_t, nullptr, hr_, nullptr, DM_, DH_);

  const long long scat = (long long)2 * NE_ * (DH_ / 4);
  const int scat_blocks = (int)((scat + 255) / 256);
  k_gcn_init<<<(NT_ * DH_ + 255) / 256, 256, 0, stream>>>(gz_, bz, dinv, hz_);
  k_gcn_scatter<<<scat_blocks, 256, 0, stream>>>(gz_, hz_, esrc, edst, ewt, dinv);
  k_gcn_init<<<(NT_ * DH_ + 255) / 256, 256, 0, stream>>>(gr_, br, dinv, hr_);
  k_gcn_scatter<<<scat_blocks, 256, 0, stream>>>(gr_, hr_, esrc, edst, ewt, dinv);
  k_sigmoid<<<(NT_ * DH_ + 255) / 256, 256, 0, stream>>>(gz_, NT_ * DH_);
  k_sigmoid<<<(NT_ * DH_ + 255) / 256, 256, 0, stream>>>(gr_, NT_ * DH_);

  // 9. candidate state: cat = [xx, zg*state] -> Wu -> GCN -> tanh
  k_cat<<<(NT_ * 192 + 255) / 256, 256, 0, stream>>>(cat_b, xx_, gz_, state);
  k_gemm<false, false, false><<<g64, 32, 0, stream>>>(cat_b, Wu_t, nullptr, hu_, nullptr, 192, DH_);
  k_gcn_init<<<(NT_ * DH_ + 255) / 256, 256, 0, stream>>>(gu_, bu, dinv, hu_);
  k_gcn_scatter<<<scat_blocks, 256, 0, stream>>>(gu_, hu_, esrc, edst, ewt, dinv);

  // 10. h = rg*st + (1-rg)*tanh(hc)
  k_final<<<(NT_ * DH_ + 255) / 256, 256, 0, stream>>>(out_h, gr_, state, gu_);
}
